// BiLstmCrfTagger_48352741818696
// MI455X (gfx1250) — compile-verified, matrix operand order checked by reading, not run on hardware
//
#include <hip/hip_runtime.h>
#include <hip/hip_bf16.h>

#define BB    128
#define SS    512
#define EMBD  256
#define HIDD  256
#define GG    1024   // 4*HID
#define NTAGS 20
#define TPAD  32
#define F2    512    // 2*HID

typedef __bf16 bf16;
typedef __attribute__((ext_vector_type(16))) __bf16 v16bf;
typedef __attribute__((ext_vector_type(8)))  __bf16 v8bf;
typedef __attribute__((ext_vector_type(8)))  float  v8f;

__device__ __forceinline__ v8f wmma_bf16(v16bf a, v16bf b, v8f c) {
  // D(f32 16x16) = A(16x32 bf16) * B(32x16 bf16) + C
  return __builtin_amdgcn_wmma_f32_16x16x32_bf16(false, a, false, b, (short)0, c,
                                                 false, false);
}

// Position of element (row/col r in 0..15, k in 0..K-1) inside a packed fragment
// buffer laid out [kstep][lane(32)][16] matching the CDNA5 16-bit A/B operand
// VGPR layout (lanes 0-15: K 0-7 & 16-23; lanes 16-31: K 8-15 & 24-31).
__device__ __forceinline__ int frag_pos(int r, int k) {
  int kstep = k >> 5, kin = k & 31, khalf, j;
  if (kin < 16) { khalf = kin >> 3; j = kin & 7; }
  else          { kin -= 16; khalf = kin >> 3; j = 8 + (kin & 7); }
  return (kstep * 32 + khalf * 16 + r) * 16 + j;
}

// ---------------------------------------------------------------------------
// Pack a row-major f32 weight [N][K] (used as B: out = in * W^T) into bf16
// WMMA B-fragment layout [K/32][N/16][32][16]. realN < N pads with zeros.
// ---------------------------------------------------------------------------
__global__ void pack_b_kernel(const float* __restrict__ W, bf16* __restrict__ dst,
                              int N, int K, int realN) {
  int idx = blockIdx.x * blockDim.x + threadIdx.x;
  int total = N * K;
  if (idx >= total) return;
  int j    = idx & 15;
  int lane = (idx >> 4) & 31;
  int nt   = (idx >> 9) % (N >> 4);
  int ks   = idx / (512 * (N >> 4));
  int col  = nt * 16 + (lane & 15);
  int khalf = lane >> 4;
  int k    = ks * 32 + khalf * 8 + (j < 8 ? j : j + 8);
  float v = (col < realN) ? W[col * K + k] : 0.0f;
  dst[idx] = (bf16)v;
}

// ---------------------------------------------------------------------------
// Fused embedding-gather + input-projection + recurrent bf16-WMMA LSTM.
// grid.x = 16: blockIdx>>3 = direction, blockIdx&7 = batch tile of 16 rows.
// 512 threads = 16 waves (4/SIMD); wave w owns gate columns [64w, 64w+64).
// Both GEMMs (x@Wih^T and h@Whh^T) are flattened into one 64-step fragment
// stream, software-pipelined with a depth-4 B buffer (8 b128 loads in flight)
// and one-k-ahead A preload. The asm on the per-lane byte offset keeps the
// loads inside the time loop (no LICM -> no spill) while preserving global
// addressing (global_load_b128 saddr + voffset + imm, never FLAT/DScnt).
// ---------------------------------------------------------------------------
__global__ __launch_bounds__(512) void lstm_kernel(
    const int* __restrict__ xs, const float* __restrict__ emb,
    const bf16* __restrict__ WihF, const bf16* __restrict__ WhhF,
    const float* __restrict__ bF,
    const bf16* __restrict__ WihR, const bf16* __restrict__ WhhR,
    const float* __restrict__ bR,
    bf16* __restrict__ feats)
{
  __shared__ __align__(32) bf16     xfrag[8 * 32 * 16];   // 8 KB
  __shared__ __align__(32) bf16     hfrag[8 * 32 * 16];   // 8 KB
  __shared__ __align__(16) _Float16 gates[16 * GG];       // 32 KB

  const int tid  = threadIdx.x;
  const int lane = tid & 31;
  const int w    = tid >> 5;          // wave 0..15
  const int dir  = blockIdx.x >> 3;
  const int r0   = (blockIdx.x & 7) * 16;

  const bf16* __restrict__ Wih = dir ? WihR : WihF;
  const bf16* __restrict__ Whh = dir ? WhhR : WhhF;
  const float* bv = dir ? bR : bF;

  // Per-lane byte offset of this wave's (k=0, nt=0) fragment.
  int wob = (((w * 4) * 32 + lane) * 16) * (int)sizeof(bf16);

  // Per-lane column bias (column index invariant across steps)
  float bias_r[4];
#pragma unroll
  for (int nt = 0; nt < 4; ++nt)
    bias_r[nt] = bv[(w * 4 + nt) * 16 + (lane & 15)];

  for (int i = tid; i < 8 * 32 * 16; i += 512) hfrag[i] = (bf16)0.0f;

  // cell-state ownership: thread -> batch row tid>>5, features jbase..jbase+7
  const int row   = tid >> 5;
  const int jbase = (tid & 31) * 8;
  float c[8];
#pragma unroll
  for (int q = 0; q < 8; ++q) c[q] = 0.0f;

  for (int t = 0; t < SS; ++t) {
    const int tcur = dir ? (SS - 1 - t) : t;

    // gather embedding slice -> x fragment (bf16, WMMA A layout)
    {
      int tok = xs[(r0 + row) * SS + tcur];
      const float4* e4 = (const float4*)(emb + (size_t)tok * EMBD + jbase);
      float4 e0 = e4[0], e1 = e4[1];
      float ev[8] = {e0.x, e0.y, e0.z, e0.w, e1.x, e1.y, e1.z, e1.w};
#pragma unroll
      for (int q = 0; q < 8; ++q)
        xfrag[frag_pos(row, jbase + q)] = (bf16)ev[q];
    }
    // prefetch next step's embedding row (global_prefetch_b8)
    if (t + 1 < SS) {
      const int tnxt = dir ? (SS - 2 - t) : (t + 1);
      int tokn = xs[(r0 + row) * SS + tnxt];
      __builtin_prefetch(emb + (size_t)tokn * EMBD + jbase, 0, 1);
    }
    __syncthreads();

    // Redefine the per-lane offset each step: defeats LICM, keeps global AS.
    asm volatile("" : "+v"(wob));
    const char* ihB = (const char*)Wih + wob;
    const char* hhB = (const char*)Whh + wob;

    // Fragment stream i = 0..63: phase = i>>5 (0: x*Wih, 1: h*Whh),
    // k = (i>>2)&7, nt = i&3. acc[nt] accumulates across phase and k.
    auto loadB = [&](int i) -> v16bf {
      const char* base = (i & 32) ? hhB : ihB;
      int r = i & 31;
      return *(const v16bf*)(base + (r >> 2) * 65536 + (r & 3) * 1024);
    };
    auto loadA = [&](int kk) -> v16bf {  // kk = 0..15 (x k=0..7, h k=0..7)
      const bf16* s = (kk < 8) ? xfrag : hfrag;
      return *(const v16bf*)&s[((kk & 7) * 32 + lane) * 16];
    };

    v8f acc[4] = {};
    v16bf bbuf[4];
#pragma unroll
    for (int p = 0; p < 4; ++p) bbuf[p] = loadB(p);
    v16bf aCur = loadA(0);
    v16bf aNxt = loadA(1);

#pragma unroll
    for (int i = 0; i < 64; ++i) {
      v16bf b = bbuf[i & 3];
      if (i + 4 < 64) bbuf[i & 3] = loadB(i + 4);
      acc[i & 3] = wmma_bf16(aCur, b, acc[i & 3]);
      if ((i & 3) == 3) {
        aCur = aNxt;
        int kk = (i >> 2) + 2;
        if (kk < 16) aNxt = loadA(kk);
      }
    }

    // scatter D tiles (+bias) to gate exchange buffer
#pragma unroll
    for (int nt = 0; nt < 4; ++nt) {
      int col = (w * 4 + nt) * 16 + (lane & 15);
#pragma unroll
      for (int v = 0; v < 8; ++v) {
        int rr = (lane < 16) ? v : v + 8;
        gates[rr * GG + col] = (_Float16)(acc[nt][v] + bias_r[nt]);
      }
    }
    __syncthreads();

    // pointwise LSTM cell update, emit h (bf16) to frag + feats (one b128)
    v8bf hpack;
#pragma unroll
    for (int q = 0; q < 8; ++q) {
      int j = jbase + q;
      float gi = (float)gates[row * GG + j];
      float gf = (float)gates[row * GG + 256 + j];
      float gg = (float)gates[row * GG + 512 + j];
      float go = (float)gates[row * GG + 768 + j];
      float si = 1.0f / (1.0f + __expf(-gi));
      float sf = 1.0f / (1.0f + __expf(-gf));
      float so = 1.0f / (1.0f + __expf(-go));
      c[q] = sf * c[q] + si * tanhf(gg);
      float h = so * tanhf(c[q]);
      bf16 hb = (bf16)h;
      hfrag[frag_pos(row, j)] = hb;
      hpack[q] = hb;
    }
    *(v8bf*)&feats[((size_t)(r0 + row) * SS + tcur) * F2 + dir * HIDD + jbase] = hpack;
    // next-iteration RAW on xfrag/hfrag is covered by the barrier after gather
  }
}

// ---------------------------------------------------------------------------
// emissions[M=B*S, 32] = feats(bf16) @ fcW^T + fc_b  via WMMA; cols 20..31 pad.
// 256 threads / 8 waves, wave w handles a 16-row tile.
// ---------------------------------------------------------------------------
__global__ __launch_bounds__(256) void emissions_kernel(
    const bf16* __restrict__ feats, const bf16* __restrict__ fcWp,
    const float* __restrict__ fc_b, float* __restrict__ em)
{
  const int lane  = threadIdx.x & 31;
  const int w     = threadIdx.x >> 5;
  const int m0    = (blockIdx.x * 8 + w) * 16;
  const int arow  = m0 + (lane & 15);
  const int khalf = lane >> 4;

  v8f acc[2] = {};
  for (int ks = 0; ks < 16; ++ks) {
    const bf16* ap = feats + (size_t)arow * F2 + ks * 32 + khalf * 8;
    v8bf lo = *(const v8bf*)ap;
    v8bf hi = *(const v8bf*)(ap + 16);
    v16bf a = __builtin_shufflevector(lo, hi, 0, 1, 2, 3, 4, 5, 6, 7,
                                      8, 9, 10, 11, 12, 13, 14, 15);
#pragma unroll
    for (int nt = 0; nt < 2; ++nt) {
      v16bf bb = *(const v16bf*)&fcWp[((ks * 2 + nt) * 32 + lane) * 16];
      acc[nt] = wmma_bf16(a, bb, acc[nt]);
    }
  }
#pragma unroll
  for (int nt = 0; nt < 2; ++nt) {
    int col = nt * 16 + (lane & 15);
    float bias = (col < NTAGS) ? fc_b[col] : 0.0f;
#pragma unroll
    for (int v = 0; v < 8; ++v) {
      int rr = m0 + ((lane < 16) ? v : v + 8);
      em[(size_t)rr * TPAD + col] = acc[nt][v] + bias;
    }
  }
}

// ---------------------------------------------------------------------------
// Viterbi: one wave per batch element; lane t' owns tag t'. trans in LDS,
// backpointers u8 in workspace, lane-0 backtrack. Matches jnp.argmax tie-break.
// ---------------------------------------------------------------------------
__global__ __launch_bounds__(32) void viterbi_kernel(
    const float* __restrict__ em, const int* __restrict__ xs,
    const float* __restrict__ trans, const float* __restrict__ start_t,
    const float* __restrict__ end_t, unsigned char* __restrict__ bp,
    float* __restrict__ out)
{
  __shared__ float trans_sh[NTAGS * NTAGS];
  __shared__ float score_sh[32];
  const int b = blockIdx.x;
  const int t = threadIdx.x;
  for (int i = t; i < NTAGS * NTAGS; i += 32) trans_sh[i] = trans[i];

  float score = (t < NTAGS) ? start_t[t] + em[((size_t)b * SS) * TPAD + t] : -1e30f;
  __syncthreads();

  for (int s = 1; s < SS; ++s) {
    score_sh[t] = score;
    __syncthreads();
    float best = -1e30f; int arg = 0;
    if (t < NTAGS) {
      for (int p = 0; p < NTAGS; ++p) {
        float v = score_sh[p] + trans_sh[p * NTAGS + t];
        if (v > best) { best = v; arg = p; }
      }
    }
    bool m = xs[b * SS + s] > 0;
    if (t < NTAGS) {
      float emv = em[((size_t)b * SS + s) * TPAD + t];
      int bpv = t;
      if (m) { score = best + emv; bpv = arg; }
      bp[((size_t)b * SS + s) * NTAGS + t] = (unsigned char)bpv;
    }
    __syncthreads();
  }

  if (t < NTAGS) score += end_t[t];
  score_sh[t] = (t < NTAGS) ? score : -1e30f;
  __syncthreads();
  if (t == 0) {
    float best = score_sh[0]; int last = 0;
    for (int p = 1; p < NTAGS; ++p)
      if (score_sh[p] > best) { best = score_sh[p]; last = p; }
    out[b] = best;
    float* tags = out + BB;
    int tag = last;
    tags[b * SS + (SS - 1)] = (float)tag;
    for (int s = SS - 1; s >= 1; --s) {
      tag = bp[((size_t)b * SS + s) * NTAGS + tag];
      tags[b * SS + (s - 1)] = (float)tag;
    }
  }
}

// ---------------------------------------------------------------------------
extern "C" void kernel_launch(void* const* d_in, const int* in_sizes, int n_in,
                              void* d_out, int out_size, void* d_ws, size_t ws_size,
                              hipStream_t stream) {
  (void)in_sizes; (void)n_in; (void)out_size; (void)ws_size;
  const int*   xs     = (const int*)d_in[0];
  const float* emb    = (const float*)d_in[1];
  const float* WihF   = (const float*)d_in[2];
  const float* WhhF   = (const float*)d_in[3];
  const float* bF     = (const float*)d_in[4];
  const float* WihR   = (const float*)d_in[5];
  const float* WhhR   = (const float*)d_in[6];
  const float* bR     = (const float*)d_in[7];
  const float* fcW    = (const float*)d_in[8];
  const float* fcb    = (const float*)d_in[9];
  const float* trans  = (const float*)d_in[10];
  const float* startt = (const float*)d_in[11];
  const float* endt   = (const float*)d_in[12];

  char* ws = (char*)d_ws;
  size_t off = 0;
  auto carve = [&](size_t bytes) {
    char* p = ws + off;
    off += (bytes + 255) & ~(size_t)255;
    return p;
  };
  bf16* WihFp = (bf16*)carve((size_t)GG * EMBD * sizeof(bf16));     // 512 KB
  bf16* WihRp = (bf16*)carve((size_t)GG * EMBD * sizeof(bf16));
  bf16* WhhFp = (bf16*)carve((size_t)GG * HIDD * sizeof(bf16));
  bf16* WhhRp = (bf16*)carve((size_t)GG * HIDD * sizeof(bf16));
  bf16* fcWp  = (bf16*)carve((size_t)TPAD * F2 * sizeof(bf16));     // 32 KB
  bf16* featsp = (bf16*)carve((size_t)BB * SS * F2 * sizeof(bf16)); // 64 MB
  float* emp   = (float*)carve((size_t)BB * SS * TPAD * sizeof(float)); // 8.4 MB
  unsigned char* bpp = (unsigned char*)carve((size_t)BB * SS * NTAGS);  // 1.3 MB

  // 1) pack weights into bf16 WMMA fragment layout
  pack_b_kernel<<<(GG * EMBD + 255) / 256, 256, 0, stream>>>(WihF, WihFp, GG, EMBD, GG);
  pack_b_kernel<<<(GG * EMBD + 255) / 256, 256, 0, stream>>>(WihR, WihRp, GG, EMBD, GG);
  pack_b_kernel<<<(GG * HIDD + 255) / 256, 256, 0, stream>>>(WhhF, WhhFp, GG, HIDD, GG);
  pack_b_kernel<<<(GG * HIDD + 255) / 256, 256, 0, stream>>>(WhhR, WhhRp, GG, HIDD, GG);
  pack_b_kernel<<<(TPAD * F2 + 255) / 256, 256, 0, stream>>>(fcW, fcWp, TPAD, F2, NTAGS);

  // 2) fused bidirectional LSTM (8 batch tiles x 2 directions, 16 waves each)
  lstm_kernel<<<16, 512, 0, stream>>>(xs, emb, WihFp, WhhFp, bF,
                                      WihRp, WhhRp, bR, featsp);

  // 3) emissions GEMM (65536 x 32 over K=512)
  emissions_kernel<<<(BB * SS) / 128, 256, 0, stream>>>(featsp, fcWp, fcb, emp);

  // 4) Viterbi decode + backtrack -> scores[128], tags[128*512]
  viterbi_kernel<<<BB, 32, 0, stream>>>(emp, xs, trans, startt, endt, bpp,
                                        (float*)d_out);
}
